// MultiHeadAttention_14070312861877
// MI455X (gfx1250) — matrix operand order repaired and checked by reading, not compiled, not run on hardware
//
#include <hip/hip_runtime.h>
#include <float.h>

#define NHEAD 8
#define SLEN  1024
#define BSZ   8
#define UD    1024
#define HD    128
#define KT    32        // keys per tile
#define NT    (SLEN / KT)
#define LDK_S 136       // ldsK row stride in halves (128 + 8 pad -> conflict-free b128)
#define LDT_S 40        // ldsT / wtile row stride in halves (32 + 8 pad)

typedef __attribute__((ext_vector_type(16))) _Float16 v16h;
typedef __attribute__((ext_vector_type(2)))  _Float16 h2f;
typedef __attribute__((ext_vector_type(8)))  float    v8f;
typedef __attribute__((ext_vector_type(4)))  unsigned v4u;
typedef __attribute__((ext_vector_type(8)))  unsigned v8u;

union Frag16 {            // one WMMA f16 operand = 16 halves = 8 VGPRs
  v16h  h;
  uint4 q[2];
};

__device__ __forceinline__ v8f vzero8() {
  v8f z;
#pragma unroll
  for (int i = 0; i < 8; ++i) z[i] = 0.0f;
  return z;
}

__device__ __forceinline__ v8f wmma16(v16h a, v16h b, v8f c) {
  return __builtin_amdgcn_wmma_f32_16x16x32_f16(false, a, false, b, (short)0, c,
                                                false, false);
}

// Issue one TDM transfer: 32 rows x 128 f32, global row stride 8192 floats,
// into LDS at byte offset ldsAddr.  Wave-uniform; caller guards to one wave.
__device__ __forceinline__ void tdm_issue(const float* gptr, unsigned ldsAddr) {
  size_t gaddr = (size_t)gptr;
  v4u g0;
  g0[0] = 1u;                                      // count = 1 valid D#
  g0[1] = ldsAddr;                                 // lds_addr
  g0[2] = (unsigned)(gaddr & 0xFFFFFFFFu);         // global_addr[31:0]
  g0[3] = (unsigned)((gaddr >> 32) & 0x01FFFFFFu)  // global_addr[56:32]
          | (2u << 30);                            // type = 2 (image)
  v8u g1;
  g1[0] = 2u << 16;              // data_size = 4 bytes
  g1[1] = 128u << 16;            // tensor_dim0 = 128
  g1[2] = (unsigned)KT << 16;    // tensor_dim1 = 32
  g1[3] = 128u << 16;            // tile_dim0 = 128
  g1[4] = (unsigned)KT;          // tile_dim1 = 32, tile_dim2 = 0
  g1[5] = (unsigned)(BSZ * UD);  // tensor_dim0_stride = 8192
  g1[6] = 0u;
  g1[7] = 0u;
  v4u gz;
  gz[0] = 0u; gz[1] = 0u; gz[2] = 0u; gz[3] = 0u;
  asm volatile("tensor_load_to_lds %0, %1, %2, %3"
               :: "s"(g0), "s"(g1), "s"(gz), "s"(gz)
               : "memory");
}

// ---------------------------------------------------------------------------
// Kernel 1: flash-attention per (batch, head).  grid = (64 head-batches, 8
// q-blocks of 128 rows).  block = 256 threads = 8 waves; wave owns 16 q rows.
// Scores are computed TRANSPOSED (M=key, N=query) so the softmax reduction is
// 16 in-lane values + one shfl_xor(16), and the resulting P tile is already in
// WMMA A-operand layout for P*V (no shuffles).  K/V tiles are staged by the
// Tensor Data Mover, double-buffered so the DMA for tile kt+1 overlaps the
// convert+compute of tile kt.
// ---------------------------------------------------------------------------
__launch_bounds__(256, 1)
__global__ void attn_fa_kernel(const float* __restrict__ x,
                               const float* __restrict__ adj,
                               _Float16* __restrict__ awork)
{
  __shared__ float    lraw[2][KT * HD];     // 2 x 16 KB raw f32 tiles (TDM dst)
  __shared__ _Float16 ldsK[KT * LDK_S];     // keys row-major  (K A-frags)
  __shared__ _Float16 ldsT[HD * LDT_S];     // d row-major / key minor (V B-frags)

  const int bh    = blockIdx.x;             // 0..63
  const int b     = bh >> 3;
  const int h     = bh & 7;
  const int qblk  = blockIdx.y;             // 0..7
  const int tid   = threadIdx.x;
  const int wave  = tid >> 5;
  const int lane  = tid & 31;
  const int l16   = lane & 15;
  const int lhalf = lane >> 4;              // 0: lanes 0-15, 1: lanes 16-31

  const float scaling = 0.08838834764831845f;  // 128^-0.5

  const int q0  = qblk * 128 + wave * 16;   // wave's first q row
  const int myq = q0 + l16;                 // this lane's query column

  // ---- persistent Q^T B-fragments: B[d,n] = Q[n,d], 4 chunks of K=32 ----
  Frag16 Qb[4];
  {
    const float* qrow = x + (size_t)myq * (BSZ * UD) + (size_t)b * UD + h * HD;
#pragma unroll
    for (int c = 0; c < 4; ++c) {
      const float* p = qrow + c * 32 + lhalf * 8;    // lanes<16: d 0-7,16-23, else 8-15,24-31
#pragma unroll
      for (int i = 0; i < 16; ++i) Qb[c].h[i] = (_Float16)p[(i & 7) + ((i >> 3) << 4)];
    }
  }

  v8f acc[8];                                // attn tile (16q x 128d), 8 C tiles
#pragma unroll
  for (int t = 0; t < 8; ++t) acc[t] = vzero8();
  float m = -1.0e30f;                        // running max (per query, lane pair)
  float l = 0.0f;                            // running sum

  const unsigned rawAddr0 = (unsigned)(size_t)(&lraw[0][0]);
  const float*   kbase    = x + (size_t)b * UD + h * HD;

  // prime the pipeline: tile 0 -> buffer 0
  if (wave == 0) tdm_issue(kbase + (size_t)0 * KT * (BSZ * UD), rawAddr0);

  for (int kt = 0; kt < NT; ++kt) {
    // ---- issue next tile, then wait for current one (in-order TDM) ----
    if (wave == 0) {
      if (kt + 1 < NT) {
        tdm_issue(kbase + (size_t)(kt + 1) * KT * (BSZ * UD),
                  rawAddr0 + ((kt + 1) & 1) * (KT * HD * 4));
        __builtin_amdgcn_s_wait_tensorcnt(1);
      } else {
        __builtin_amdgcn_s_wait_tensorcnt(0);
      }
    }
    __syncthreads();    // tile kt visible; previous ldsK/ldsT reads all done

    // ---- convert f32 -> f16, build row-major + transposed copies ----
    const float* buf = &lraw[kt & 1][0];
#pragma unroll
    for (int i = 0; i < 8; ++i) {
      int e   = (tid + i * 256) * 2;         // even element index 0..4094
      int key = e >> 7;
      int d   = e & 127;                     // even, pair stays in row
      float2 f = *(const float2*)(buf + e);
      h2f hp;
      hp[0] = (_Float16)f.x;
      hp[1] = (_Float16)f.y;
      *(h2f*)&ldsK[key * LDK_S + d] = hp;
      ldsT[d * LDT_S + key]       = hp[0];
      ldsT[(d + 1) * LDT_S + key] = hp[1];
    }
    __syncthreads();                         // f16 tiles ready

    // ---- adjacency mask rows (global, issued early) ----
    const float* arow = adj + ((size_t)b << 20) + (size_t)myq * SLEN
                            + kt * KT + lhalf * 8;
    float4 aj0 = ((const float4*)arow)[0];
    float4 aj1 = ((const float4*)arow)[1];
    float4 aj2 = ((const float4*)(arow + 16))[0];
    float4 aj3 = ((const float4*)(arow + 16))[1];
    if (kt + 1 < NT) __builtin_prefetch(arow + KT, 0, 0);

    // ---- preload all K A-fragments (one clause of ds_load_b128) ----
    Frag16 ka0[4], ka1[4];
#pragma unroll
    for (int c = 0; c < 4; ++c) {
      const _Float16* p0 = &ldsK[l16 * LDK_S        + c * 32 + lhalf * 8];
      const _Float16* p1 = &ldsK[(16 + l16) * LDK_S + c * 32 + lhalf * 8];
      ka0[c].q[0] = *(const uint4*)(p0);
      ka0[c].q[1] = *(const uint4*)(p0 + 16);
      ka1[c].q[0] = *(const uint4*)(p1);
      ka1[c].q[1] = *(const uint4*)(p1 + 16);
    }

    // ---- S^T = K * Q^T : two 16x16 tiles (key halves), K-dim = 128 ----
    v8f s0 = vzero8(), s1 = vzero8();
#pragma unroll
    for (int c = 0; c < 4; ++c) {
      s0 = wmma16(ka0[c].h, Qb[c].h, s0);
      s1 = wmma16(ka1[c].h, Qb[c].h, s1);
    }

    // ---- additive mask (faithful to reference) + online softmax ----
    float ajA[8] = {aj0.x, aj0.y, aj0.z, aj0.w, aj1.x, aj1.y, aj1.z, aj1.w};
    float ajB[8] = {aj2.x, aj2.y, aj2.z, aj2.w, aj3.x, aj3.y, aj3.z, aj3.w};
    float sv[16];
    float mloc = -3.0e38f;
#pragma unroll
    for (int i = 0; i < 8; ++i) {
      sv[i]     = fmaf(s0[i], scaling, (ajA[i] - 1.0f) * 1.0e30f);
      sv[8 + i] = fmaf(s1[i], scaling, (ajB[i] - 1.0f) * 1.0e30f);
      mloc = fmaxf(mloc, fmaxf(sv[i], sv[8 + i]));
    }
    mloc = fmaxf(mloc, __shfl_xor(mloc, 16, 32));
    float mnew = fmaxf(m, mloc);
    float corr = __expf(m - mnew);
    float psum = 0.0f;
    float p[16];
#pragma unroll
    for (int i = 0; i < 16; ++i) {
      p[i] = __expf(sv[i] - mnew);           // masked entries underflow to 0
      psum += p[i];
    }
    psum += __shfl_xor(psum, 16, 32);
    l = l * corr + psum;
    m = mnew;

    // ---- rescale accumulators: row M = v + 8*lhalf needs lane M's corr ----
    float cr[8];
#pragma unroll
    for (int v = 0; v < 8; ++v) cr[v] = __shfl(corr, v + 8 * lhalf, 32);
#pragma unroll
    for (int t = 0; t < 8; ++t)
#pragma unroll
      for (int v = 0; v < 8; ++v) acc[t][v] *= cr[v];

    // ---- P tile is already in A-operand layout: just pack f32 -> f16 ----
    Frag16 pf;
#pragma unroll
    for (int i = 0; i < 16; ++i) pf.h[i] = (_Float16)p[i];

    // ---- preload all V B-fragments, then acc += P * V (8 d-tiles) ----
    Frag16 vb[8];
#pragma unroll
    for (int t = 0; t < 8; ++t) {
      const _Float16* vbase = &ldsT[(16 * t + l16) * LDT_S + lhalf * 8];
      vb[t].q[0] = *(const uint4*)(vbase);
      vb[t].q[1] = *(const uint4*)(vbase + 16);
    }
#pragma unroll
    for (int t = 0; t < 8; ++t) acc[t] = wmma16(pf.h, vb[t].h, acc[t]);
  }

  // ---- epilogue: divide by row sum, store f16 to A[s*8+b, h*128+d] ----
  float linv = 1.0f / l;
  float rs[8];
#pragma unroll
  for (int v = 0; v < 8; ++v) rs[v] = __shfl(linv, v + 8 * lhalf, 32);
#pragma unroll
  for (int t = 0; t < 8; ++t) {
#pragma unroll
    for (int v = 0; v < 8; ++v) {
      int srow = q0 + v + 8 * lhalf;                 // global sequence index
      int col  = h * HD + 16 * t + l16;              // h*128 + d
      awork[(size_t)(srow * BSZ + b) * UD + col] = (_Float16)(acc[t][v] * rs[v]);
    }
  }
}

// ---------------------------------------------------------------------------
// Kernel 2: out(8192x1024, f32) = A(8192x1024, f16) @ W^T  (W = o_w, f32).
// grid = (64 row-blocks, 8 col-blocks); block = 256 thr; wave: 16 rows x 128
// cols.  W tile (128 cols x 32 k) converted to f16 in LDS, shared by 8 waves.
// ---------------------------------------------------------------------------
__launch_bounds__(256, 1)
__global__ void proj_gemm_kernel(const _Float16* __restrict__ A,
                                 const float* __restrict__ W,
                                 float* __restrict__ out)
{
  __shared__ _Float16 wt[128 * LDT_S];       // 10 KB

  const int rb    = blockIdx.x;              // 0..63
  const int cb    = blockIdx.y;              // 0..7
  const int tid   = threadIdx.x;
  const int wave  = tid >> 5;
  const int lane  = tid & 31;
  const int l16   = lane & 15;
  const int lhalf = lane >> 4;

  const int row0 = rb * 128 + wave * 16;

  v8f acc[8];
#pragma unroll
  for (int t = 0; t < 8; ++t) acc[t] = vzero8();

  for (int kc = 0; kc < UD / 32; ++kc) {
    __syncthreads();
    // stage W tile: cols [cb*128, +128), k [kc*32, +32) ; f32 -> f16 (pairs)
#pragma unroll
    for (int i = 0; i < 8; ++i) {
      int e = (tid + i * 256) * 2;           // even element, 0..4094
      int c = e >> 5;
      int k = e & 31;                        // even
      float2 f = *(const float2*)&W[(size_t)(cb * 128 + c) * UD + kc * 32 + k];
      h2f hp;
      hp[0] = (_Float16)f.x;
      hp[1] = (_Float16)f.y;
      *(h2f*)&wt[c * LDT_S + k] = hp;
    }
    // prefetch next W tile slice (global_prefetch_b8)
    if (kc + 1 < UD / 32) {
      int c = tid >> 1;
      int k = (tid & 1) * 16;
      __builtin_prefetch(&W[(size_t)(cb * 128 + c) * UD + (kc + 1) * 32 + k], 0, 0);
    }
    __syncthreads();

    // A fragment: 16 rows x 32 k, straight from f16 workspace (row-major)
    Frag16 af;
    const _Float16* abase = A + (size_t)(row0 + l16) * UD + kc * 32 + lhalf * 8;
    af.q[0] = *(const uint4*)(abase);
    af.q[1] = *(const uint4*)(abase + 16);

    // preload all 8 B fragments, then run the WMMA chain back-to-back
    Frag16 bf[8];
#pragma unroll
    for (int t = 0; t < 8; ++t) {
      const _Float16* bbase = &wt[(16 * t + l16) * LDT_S + lhalf * 8];
      bf[t].q[0] = *(const uint4*)(bbase);
      bf[t].q[1] = *(const uint4*)(bbase + 16);
    }
#pragma unroll
    for (int t = 0; t < 8; ++t) acc[t] = wmma16(af.h, bf[t].h, acc[t]);
  }

#pragma unroll
  for (int t = 0; t < 8; ++t) {
#pragma unroll
    for (int v = 0; v < 8; ++v) {
      int r = row0 + v + 8 * lhalf;
      int c = cb * 128 + 16 * t + l16;
      out[(size_t)r * UD + c] = acc[t][v];
    }
  }
}

extern "C" void kernel_launch(void* const* d_in, const int* in_sizes, int n_in,
                              void* d_out, int out_size, void* d_ws, size_t ws_size,
                              hipStream_t stream) {
  const float* x   = (const float*)d_in[0];   // [1024, 8, 1024]
  const float* adj = (const float*)d_in[1];   // [8, 1024, 1024]
  const float* o_w = (const float*)d_in[2];   // [1024, 1024]
  float*       out = (float*)d_out;           // [8, 1024, 1024] flat
  _Float16*    aw  = (_Float16*)d_ws;         // [8192, 1024] f16 (16 MB)

  attn_fa_kernel<<<dim3(64, 8), 256, 0, stream>>>(x, adj, aw);
  proj_gemm_kernel<<<dim3(64, 8), 256, 0, stream>>>(aw, o_w, out);
}
